// EfficientMLA_36292473651445
// MI455X (gfx1250) — matrix operand order repaired and checked by reading, not compiled, not run on hardware
//
#include <hip/hip_runtime.h>
#include <hip/hip_bf16.h>

#define DM   4096      // dim_model
#define LOWD 512       // low_dim
#define SP   4096      // past length
#define SEQ  4097      // total kv length
#define NB   4         // batch

#define USE_TDM 1      // Tensor Data Mover for B-tile staging (fallback: VGPR copy)

typedef __attribute__((ext_vector_type(16))) __bf16 v16bf;
typedef __attribute__((ext_vector_type(8)))  float  v8f;
typedef __attribute__((ext_vector_type(4)))  int    v4i;
typedef __attribute__((ext_vector_type(8)))  int    v8i;

union Frag {
    v16bf v;
    uint4 q[2];
    unsigned short h[16];
};

__device__ __forceinline__ unsigned short f2bf(float f) {
    unsigned u = __float_as_uint(f);
    u += 0x7FFFu + ((u >> 16) & 1u);      // round-to-nearest-even
    return (unsigned short)(u >> 16);
}
__device__ __forceinline__ float bf2f(unsigned short h) {
    return __uint_as_float(((unsigned)h) << 16);
}

#if USE_TDM
// Issue a TDM 2D tile load: 32 rows x 512 bf16 cols from a [4096 x 512] bf16
// tensor (row stride 512 elements) into LDS at lds_addr. Tracked by TENSORcnt.
__device__ __forceinline__ void tdm_load_b_tile(const unsigned short* gptr,
                                                unsigned lds_addr) {
    unsigned long long ga = (unsigned long long)gptr;
    unsigned ga_lo = __builtin_amdgcn_readfirstlane((unsigned)(ga & 0xFFFFFFFFull));
    unsigned ga_hi = __builtin_amdgcn_readfirstlane((unsigned)(ga >> 32));
    unsigned lds   = __builtin_amdgcn_readfirstlane(lds_addr);
    v4i g0;
    g0.x = 1;                                          // count=1, user descriptor
    g0.y = (int)lds;                                   // lds_addr (bytes)
    g0.z = (int)ga_lo;                                 // global_addr[31:0]
    g0.w = (int)((ga_hi & 0x01FFFFFFu) | 0x80000000u); // global_addr[56:32] | type=2
    v8i g1;
    g1[0] = 0x00010000;                         // data_size=1 (2 bytes), no mask/pad
    g1[1] = (int)(((unsigned)LOWD & 0xFFFFu) << 16);   // tensor_dim0 = 512 (lo16)
    g1[2] = (int)(((unsigned)DM   & 0xFFFFu) << 16);   // dim0 hi=0 | tensor_dim1=4096 lo
    g1[3] = (int)(((unsigned)LOWD) << 16);             // dim1 hi=0 | tile_dim0=512
    g1[4] = 32;                                        // tile_dim1=32, tile_dim2=0
    g1[5] = LOWD;                                      // tensor_dim0_stride = 512
    g1[6] = 0;                                         // stride hi | dim1_stride lo
    g1[7] = 0;                                         // dim1_stride hi
    asm volatile("tensor_load_to_lds %0, %1" :: "s"(g0), "s"(g1) : "memory");
}
#endif

// ---------------------------------------------------------------------------
// Zero-fill
__global__ __launch_bounds__(256) void kz(float* __restrict__ p, int n) {
    int i = blockIdx.x * 256 + threadIdx.x;
    if (i < n) p[i] = 0.f;
}

// fp32 -> bf16 convert (Wdkv)
__global__ __launch_bounds__(256) void kcvt(const float* __restrict__ w,
                                            unsigned short* __restrict__ o, int n) {
    int i = blockIdx.x * 256 + threadIdx.x;
    if (i < n) o[i] = f2bf(w[i]);
}

// ---------------------------------------------------------------------------
// Main fused kernel: ckv_bf16[b,s,:] = bf16(X[b,s,:]) @ Wdkv_bf16  (WMMA)
//                    scores[b,s]     = X[b,s,:] . qr_eff[b,:]      (fused dot)
// One block = 32 rows (s) x all 512 cols, full K=4096. X row read ONCE.
// Double-buffered LDS; B tile streamed by the Tensor Data Mover.
__global__ __launch_bounds__(256) void ka_gemm(
    const float* __restrict__ past_x, const float* __restrict__ x,
    const unsigned short* __restrict__ wdkv_bf,
    const float* __restrict__ qr_eff,
    unsigned short* __restrict__ ckv_bf,
    float* __restrict__ scores)
{
    __shared__ unsigned short Als[2][32][32];    // A tiles (M=32, K=32) bf16
    __shared__ unsigned short Bls[2][32][LOWD];  // B tiles (K=32, N=512) bf16
    __shared__ float qr_lds[DM];                 // absorbed rope-query, this batch
    __shared__ float sred[32][8];                // score_r partial reduction

    const int tid  = threadIdx.x;
    const int lane = tid & 31;
    const int wave = tid >> 5;                   // 0..7 -> 64-wide N slice
    const int b    = blockIdx.y;
    const int s0   = blockIdx.x * 32;

    for (int i = tid; i < DM; i += 256) qr_lds[i] = qr_eff[b * DM + i];

    v8f acc[2][4];
    const v8f vz = {0.f,0.f,0.f,0.f,0.f,0.f,0.f,0.f};
    #pragma unroll
    for (int i = 0; i < 2; ++i)
        #pragma unroll
        for (int j = 0; j < 4; ++j) acc[i][j] = vz;

    // A-tile staging: each thread owns one float4 (row ar, cols ac..ac+3)
    const int ar = tid >> 3;
    const int ac = (tid & 7) * 4;
    int s = s0 + ar;
    int sc = (s < SEQ) ? s : (SEQ - 1);
    const float* arow = (sc < SP) ? (past_x + ((size_t)b * SP + sc) * DM)
                                  : (x + (size_t)b * DM);

    // fused score_r dot assignment: 8 threads per row, 4 k's each
    const int dr = tid & 31;
    const int dk = (tid >> 5) * 4;
    float sacc = 0.f;

#if USE_TDM
    const unsigned ldsB0 =
        __builtin_amdgcn_readfirstlane((unsigned)(size_t)&Bls[0][0][0]);
    const unsigned ldsB1 =
        __builtin_amdgcn_readfirstlane((unsigned)(size_t)&Bls[1][0][0]);
#endif

    // ---- prologue: stage A tile 0, kick TDM for B tile 0 ----
    {
        float4 av = *(const float4*)(arow + ac);
        uint2 ap;
        ap.x = (unsigned)f2bf(av.x) | ((unsigned)f2bf(av.y) << 16);
        ap.y = (unsigned)f2bf(av.z) | ((unsigned)f2bf(av.w) << 16);
        *(uint2*)&Als[0][ar][ac] = ap;
    }
#if USE_TDM
    if (wave == 0) tdm_load_b_tile(wdkv_bf, ldsB0);
#else
    #pragma unroll
    for (int i = 0; i < 8; ++i) {
        int idx = (i * 256 + tid) * 8;
        *(uint4*)&Bls[0][idx >> 9][idx & 511] =
            *(const uint4*)(wdkv_bf + (size_t)(idx >> 9) * LOWD + (idx & 511));
    }
#endif

    const int NS = DM / 32;
    for (int i = 0; i < NS; ++i) {
        const int buf = i & 1;
#if USE_TDM
        if (wave == 0) __builtin_amdgcn_s_wait_tensorcnt(0);  // B[buf] in LDS
#endif
        __syncthreads();

        const bool hasNext = (i + 1 < NS);
        float4 av;
        if (hasNext) av = *(const float4*)(arow + (i + 1) * 32 + ac);

        const int k0 = i * 32;
        // fused score_r partial (reads A tile already in LDS)
        #pragma unroll
        for (int q2 = 0; q2 < 4; ++q2)
            sacc += bf2f(Als[buf][dr][dk + q2]) * qr_lds[k0 + dk + q2];

        // A fragments: 16x32 bf16 per ISA layout
        Frag afr[2];
        #pragma unroll
        for (int tm = 0; tm < 2; ++tm) {
            int row = tm * 16 + (lane & 15);
            int kb  = (lane >> 4) * 8;
            afr[tm].q[0] = *(const uint4*)&Als[buf][row][kb];      // K = kb..kb+7
            afr[tm].q[1] = *(const uint4*)&Als[buf][row][16 + kb]; // K = 16+kb..
        }
        // B fragments: 32x16 bf16, lane%16 = N, elements over K in table order (matches A)
        #pragma unroll
        for (int tn = 0; tn < 4; ++tn) {
            Frag bfr;
            int n0 = wave * 64 + tn * 16;
            int bn = n0 + (lane & 15);
            int bk = (lane >> 4) * 8;
            #pragma unroll
            for (int e = 0; e < 8; ++e) {
                bfr.h[e]     = Bls[buf][bk + e][bn];
                bfr.h[8 + e] = Bls[buf][16 + bk + e][bn];
            }
            #pragma unroll
            for (int tm = 0; tm < 2; ++tm)
                acc[tm][tn] = __builtin_amdgcn_wmma_f32_16x16x32_bf16(
                    false, afr[tm].v, false, bfr.v,
                    (short)0, acc[tm][tn], false, false);
        }

        // stage next tiles into the other buffer (overlaps with this step's math)
        if (hasNext) {
            uint2 ap;
            ap.x = (unsigned)f2bf(av.x) | ((unsigned)f2bf(av.y) << 16);
            ap.y = (unsigned)f2bf(av.z) | ((unsigned)f2bf(av.w) << 16);
            *(uint2*)&Als[buf ^ 1][ar][ac] = ap;
#if USE_TDM
            if (wave == 0)
                tdm_load_b_tile(wdkv_bf + (size_t)(i + 1) * 32 * LOWD,
                                buf ? ldsB0 : ldsB1);
#else
            #pragma unroll
            for (int q2 = 0; q2 < 8; ++q2) {
                int idx = (q2 * 256 + tid) * 8;
                *(uint4*)&Bls[buf ^ 1][idx >> 9][idx & 511] =
                    *(const uint4*)(wdkv_bf +
                                    (size_t)(i + 1) * 32 * LOWD +
                                    (size_t)(idx >> 9) * LOWD + (idx & 511));
            }
#endif
        }
    }

    // reduce + write fused score_r
    sred[dr][tid >> 5] = sacc;
    __syncthreads();
    if (tid < 32) {
        float t = 0.f;
        #pragma unroll
        for (int g = 0; g < 8; ++g) t += sred[tid][g];
        int ss = s0 + tid;
        if (ss < SEQ) scores[(size_t)b * SEQ + ss] = t;
    }

    // store ckv as bf16 (D layout: VGPR e -> M=e (lanes 0-15) / e+8 (lanes 16-31))
    #pragma unroll
    for (int tm = 0; tm < 2; ++tm)
        #pragma unroll
        for (int tn = 0; tn < 4; ++tn)
            #pragma unroll
            for (int e = 0; e < 8; ++e) {
                int row = tm * 16 + e + ((lane >> 4) ? 8 : 0);
                int col = wave * 64 + tn * 16 + (lane & 15);
                int ss = s0 + row;
                if (ss < SEQ)
                    ckv_bf[((size_t)b * SEQ + ss) * LOWD + col] = f2bf(acc[tm][tn][e]);
            }
}

// ---------------------------------------------------------------------------
// Column-accumulate GEMV: Y[b][j] += sum_{d in chunk} X[b][d] * W[d*ldw+colOff+j]
__global__ __launch_bounds__(256) void kcol(
    const float* __restrict__ X, const float* __restrict__ W,
    float* __restrict__ Y, int Din, int Dout, int ldw, int colOff, int dchunk)
{
    int j = blockIdx.x * 256 + threadIdx.x;
    if (j >= Dout) return;
    int d0 = blockIdx.y * dchunk;
    int d1 = d0 + dchunk; if (d1 > Din) d1 = Din;
    float a0 = 0.f, a1 = 0.f, a2 = 0.f, a3 = 0.f;
    for (int d = d0; d < d1; ++d) {
        float wv = W[(size_t)d * ldw + colOff + j];
        a0 += X[0 * Din + d] * wv;
        a1 += X[1 * Din + d] * wv;
        a2 += X[2 * Din + d] * wv;
        a3 += X[3 * Din + d] * wv;
    }
    atomicAdd(&Y[0 * Dout + j], a0);
    atomicAdd(&Y[1 * Dout + j], a1);
    atomicAdd(&Y[2 * Dout + j], a2);
    atomicAdd(&Y[3 * Dout + j], a3);
}

// Row-dot GEMV: Y[b][row] = sum_j W[row*ldw+j] * V[b][j]   (one wave32 per row)
__global__ __launch_bounds__(256) void krow(
    const float* __restrict__ W, const float* __restrict__ V,
    float* __restrict__ Y, int nrows, int len, int ldw)
{
    int wave = threadIdx.x >> 5, lane = threadIdx.x & 31;
    int row = blockIdx.x * 8 + wave;
    if (row >= nrows) return;
    const float* wr = W + (size_t)row * ldw;
    float a0 = 0.f, a1 = 0.f, a2 = 0.f, a3 = 0.f;
    for (int j = lane; j < len; j += 32) {
        float wv = wr[j];
        a0 += wv * V[0 * len + j];
        a1 += wv * V[1 * len + j];
        a2 += wv * V[2 * len + j];
        a3 += wv * V[3 * len + j];
    }
    #pragma unroll
    for (int o = 16; o; o >>= 1) {
        a0 += __shfl_xor(a0, o);
        a1 += __shfl_xor(a1, o);
        a2 += __shfl_xor(a2, o);
        a3 += __shfl_xor(a3, o);
    }
    if (lane == 0) {
        Y[0 * nrows + row] = a0;
        Y[1 * nrows + row] = a1;
        Y[2 * nrows + row] = a2;
        Y[3 * nrows + row] = a3;
    }
}

// scores[b,s] += qc_eff[b] . ckv_bf[b,s]   (one wave per (b,s))
__global__ __launch_bounds__(256) void ks_up(
    const unsigned short* __restrict__ ckv, const float* __restrict__ qc_eff,
    float* __restrict__ scores)
{
    int wave = threadIdx.x >> 5, lane = threadIdx.x & 31;
    long m = (long)blockIdx.x * 8 + wave;
    if (m >= (long)NB * SEQ) return;
    int b = (int)(m / SEQ);
    const unsigned short* cr = ckv + (size_t)m * LOWD;
    const float* q = qc_eff + b * LOWD;
    float a = 0.f;
    for (int j = lane; j < LOWD; j += 32) a += bf2f(cr[j]) * q[j];
    #pragma unroll
    for (int o = 16; o; o >>= 1) a += __shfl_xor(a, o);
    if (lane == 0) scores[m] += a;
}

// softmax over S per batch
__global__ __launch_bounds__(256) void ksoftmax(
    const float* __restrict__ scores, float* __restrict__ w)
{
    __shared__ float red[256];
    const float scale = 0.011048543456039806f;   // 1/sqrt(2*4096)
    int b = blockIdx.x, tid = threadIdx.x;
    float mx = -1e30f;
    for (int s2 = tid; s2 < SEQ; s2 += 256)
        mx = fmaxf(mx, scores[b * SEQ + s2] * scale);
    red[tid] = mx; __syncthreads();
    for (int o = 128; o; o >>= 1) {
        if (tid < o) red[tid] = fmaxf(red[tid], red[tid + o]);
        __syncthreads();
    }
    mx = red[0]; __syncthreads();
    float sum = 0.f;
    for (int s2 = tid; s2 < SEQ; s2 += 256) {
        float e = __expf(scores[b * SEQ + s2] * scale - mx);
        w[b * SEQ + s2] = e;
        sum += e;
    }
    red[tid] = sum; __syncthreads();
    for (int o = 128; o; o >>= 1) {
        if (tid < o) red[tid] += red[tid + o];
        __syncthreads();
    }
    float inv = 1.f / red[0];
    for (int s2 = tid; s2 < SEQ; s2 += 256) w[b * SEQ + s2] *= inv;
}

// ctx[b,l] += sum_s w[b,s] * ckv_bf[b,s,l]
__global__ __launch_bounds__(256) void kctx(
    const unsigned short* __restrict__ ckv, const float* __restrict__ w,
    float* __restrict__ ctx)
{
    int b = blockIdx.y, tid = threadIdx.x;
    float a0 = 0.f, a1 = 0.f;
    for (int s2 = blockIdx.x; s2 < SEQ; s2 += 32) {
        float ww = w[b * SEQ + s2];
        const unsigned short* cr = ckv + ((size_t)b * SEQ + s2) * LOWD;
        a0 += ww * bf2f(cr[tid]);
        a1 += ww * bf2f(cr[tid + 256]);
    }
    atomicAdd(&ctx[b * LOWD + tid], a0);
    atomicAdd(&ctx[b * LOWD + tid + 256], a1);
}

// ---------------------------------------------------------------------------
extern "C" void kernel_launch(void* const* d_in, const int* in_sizes, int n_in,
                              void* d_out, int out_size, void* d_ws, size_t ws_size,
                              hipStream_t stream) {
    (void)in_sizes; (void)n_in; (void)out_size; (void)ws_size;
    const float* x      = (const float*)d_in[0];
    const float* past_x = (const float*)d_in[1];
    const float* Wdq    = (const float*)d_in[2];
    const float* Wuq    = (const float*)d_in[3];
    const float* Wqr    = (const float*)d_in[4];
    const float* Wdkv   = (const float*)d_in[5];
    const float* Wkr    = (const float*)d_in[6];
    const float* fu     = (const float*)d_in[7];
    const float* Wo     = (const float*)d_in[8];
    float* out = (float*)d_out;

    char* ws = (char*)d_ws;
    size_t off = 0;
    auto alloc = [&](size_t bytes) { size_t o = off; off += (bytes + 255) & ~(size_t)255; return o; };
    unsigned short* wdkv_bf = (unsigned short*)(ws + alloc((size_t)DM * LOWD * 2));
    unsigned short* ckv_bf  = (unsigned short*)(ws + alloc((size_t)NB * SEQ * LOWD * 2));
    float* scores = (float*)(ws + alloc((size_t)NB * SEQ * 4));
    float* wsm    = (float*)(ws + alloc((size_t)NB * SEQ * 4));
    float* Cq     = (float*)(ws + alloc((size_t)NB * DM * 4));
    float* Qc     = (float*)(ws + alloc((size_t)NB * DM * 4));
    float* Qr     = (float*)(ws + alloc((size_t)NB * DM * 4));
    float* qr_eff = (float*)(ws + alloc((size_t)NB * DM * 4));
    float* qc_eff = (float*)(ws + alloc((size_t)NB * LOWD * 4));
    float* ctx    = (float*)(ws + alloc((size_t)NB * LOWD * 4));
    float* attn   = (float*)(ws + alloc((size_t)NB * DM * 4));

    // zero atomic-accumulated buffers + output
    kz<<<64, 256, 0, stream>>>(Cq,   NB * DM);
    kz<<<64, 256, 0, stream>>>(Qc,   NB * DM);
    kz<<<64, 256, 0, stream>>>(Qr,   NB * DM);
    kz<<<64, 256, 0, stream>>>(attn, NB * DM);
    kz<<<8,  256, 0, stream>>>(ctx,  NB * LOWD);
    kz<<<64, 256, 0, stream>>>(out,  NB * DM);

    // Wdkv -> bf16 (L2-resident B operand)
    kcvt<<<(DM * LOWD) / 256, 256, 0, stream>>>(Wdkv, wdkv_bf, DM * LOWD);

    // query chain (absorbed weights)
    kcol<<<dim3(DM / 256, 8), 256, 0, stream>>>(x,  Wdq, Cq, DM, DM, DM, 0, 512);
    kcol<<<dim3(DM / 256, 8), 256, 0, stream>>>(Cq, Wuq, Qc, DM, DM, DM, 0, 512);
    kcol<<<dim3(DM / 256, 8), 256, 0, stream>>>(Cq, Wqr, Qr, DM, DM, DM, 0, 512);
    krow<<<DM / 8,   256, 0, stream>>>(Wkr, Qr, qr_eff, DM,   DM, DM);      // Wkr @ Qr
    krow<<<LOWD / 8, 256, 0, stream>>>(fu,  Qc, qc_eff, LOWD, DM, 2 * DM);  // W_kup^T @ Qc

    // main WMMA pass: ckv (bf16) + fused rope scores, past_x read once
    ka_gemm<<<dim3((SEQ + 31) / 32, NB), 256, 0, stream>>>(
        past_x, x, wdkv_bf, qr_eff, ckv_bf, scores);

    // scores += qc_eff . ckv   (ckv from L2)
    ks_up<<<(NB * SEQ + 7) / 8, 256, 0, stream>>>(ckv_bf, qc_eff, scores);

    ksoftmax<<<NB, 256, 0, stream>>>(scores, wsm);

    // ctx = sum_s w * ckv
    kctx<<<dim3(32, NB), 256, 0, stream>>>(ckv_bf, wsm, ctx);

    // attn = ctx @ Wv (second half of fused_upsample), out = attn @ Wo
    kcol<<<dim3(DM / 256, 2), 256, 0, stream>>>(ctx,  fu, attn, LOWD, DM, 2 * DM, DM, 256);
    kcol<<<dim3(DM / 256, 8), 256, 0, stream>>>(attn, Wo, out,  DM,   DM, DM,     0,  512);
}
